// DecoderGRU_2027224564104
// MI455X (gfx1250) — matrix operand order repaired and checked by reading, not compiled, not run on hardware
//
#include <hip/hip_runtime.h>
#include <hip/hip_bf16.h>

// Sizes from the reference
#define BATCH  32
#define SEQ    64
#define EMBED  512
#define HIDDEN 512
#define VOCAB  32000
#define CW     (EMBED + HIDDEN)   // 1024
#define TB     (SEQ * BATCH)      // 2048 rows of the time-major activation matrix

typedef __attribute__((ext_vector_type(16))) __bf16 v16bf;
typedef __attribute__((ext_vector_type(8)))  __bf16 v8bf;
typedef __attribute__((ext_vector_type(8)))  float  v8f;

union U16 { v16bf v; v8bf h[2]; };

// Fragment loaders for one K=32 block of a 16x16 bf16 WMMA tile.
//  A fragment: lane holds K=[kb+g*16 .. kb+g*16+15]  (g = lane>=16)
//              -> two contiguous 16B loads at ap and ap+8 halves
//  B fragment: lane holds column n=lane%16, K=[kb+g*16 .. kb+g*16+15]
//              -> 32 contiguous bytes at bp (row-major W[N,K] gives this for free)
__device__ __forceinline__ v16bf load_afrag(const __bf16* ap) {
  U16 a;
  a.h[0] = *(const v8bf*)(ap);
  a.h[1] = *(const v8bf*)(ap + 8);
  return a.v;
}
__device__ __forceinline__ v16bf load_bfrag(const __bf16* bp) {
  U16 b;
  b.h[0] = *(const v8bf*)(bp);
  b.h[1] = *(const v8bf*)(bp + 8);
  return b.v;
}
__device__ __forceinline__ v8f wmma_acc(v16bf a, v16bf b, v8f acc) {
  return __builtin_amdgcn_wmma_f32_16x16x32_bf16(false, a, false, b,
                                                 (short)0, acc, false, false);
}
__device__ __forceinline__ v8f wmma_bf16(const __bf16* ap, const __bf16* bp, v8f acc) {
  return wmma_acc(load_afrag(ap), load_bfrag(bp), acc);
}

__device__ __forceinline__ float sigmoidf_fast(float x) {
  return 1.0f / (1.0f + __expf(-x));
}

// ---------------- prep kernels (convert weights to bf16, gather embeddings) --------------

__global__ void k_prep_gru(const float* __restrict__ Wz, const float* __restrict__ Wr,
                           const float* __restrict__ Wh, __bf16* __restrict__ Wg,
                           unsigned* __restrict__ bar) {
  if (blockIdx.x == 0 && threadIdx.x == 0) { bar[0] = 0u; bar[1] = 0u; }
  size_t i = (size_t)blockIdx.x * 256 + threadIdx.x;       // 3 * 512 * 1024 elements
  const size_t per = (size_t)HIDDEN * CW;
  if (i < 3 * per) {
    const float* src = (i < per) ? Wz : (i < 2 * per ? Wr : Wh);
    Wg[i] = (__bf16)src[i % per];
  }
}

__global__ void k_prep_fc(const float* __restrict__ fc_w, __bf16* __restrict__ fcw) {
  size_t i = (size_t)blockIdx.x * 256 + threadIdx.x;       // 32000 * 512
  if (i < (size_t)VOCAB * HIDDEN) fcw[i] = (__bf16)fc_w[i];
}

__global__ void k_embed(const int* __restrict__ captions, const float* __restrict__ embedding,
                        __bf16* __restrict__ x_all) {
  size_t i = (size_t)blockIdx.x * 256 + threadIdx.x;       // 2048 * 512
  if (i < (size_t)TB * EMBED) {
    int row = (int)(i >> 9);            // t*32 + b
    int e   = (int)(i & 511);
    int t = row >> 5, b = row & 31;
    int tok = captions[b * SEQ + t];
    x_all[i] = (__bf16)embedding[(size_t)tok * EMBED + e];
  }
}

__global__ void k_h0(const float* __restrict__ features, float* __restrict__ h_f32,
                     __bf16* __restrict__ h_bf16) {
  int i = blockIdx.x * 256 + threadIdx.x;                  // 32 * 512
  if (i < BATCH * HIDDEN) { float v = features[i]; h_f32[i] = v; h_bf16[i] = (__bf16)v; }
}

// ---------------- x-part GEMM: Ax[g] = X @ Wg_x.T + b_g for g in {z,r,h} ----------------
// M = 2048, N = 512 per gate, K = 512.  2x2 tiles (32x32 output) per wave for register reuse.
// Per gate: 64 M-groups x 16 N-groups = 1024 wave jobs; 3 gates -> 3072 waves -> 384 blocks.
__global__ void k_xgemm(const __bf16* __restrict__ x_all, const __bf16* __restrict__ Wg,
                        const float* __restrict__ bz, const float* __restrict__ br,
                        const float* __restrict__ bh, float* __restrict__ Ax) {
  int lane = threadIdx.x & 31;
  int wid  = blockIdx.x * 8 + (threadIdx.x >> 5);          // 0 .. 3071
  int g    = wid >> 10;                                    // gate
  int rem  = wid & 1023;
  int mg   = rem >> 4, ng = rem & 15;                      // 32-row x 32-col group

  const __bf16* ap = x_all + (size_t)(mg * 32 + (lane & 15)) * EMBED + ((lane >> 4) * 16);
  const __bf16* bp = Wg + (size_t)g * HIDDEN * CW
                        + (size_t)(ng * 32 + (lane & 15)) * CW + ((lane >> 4) * 16); // cols 0..511 = x-part
  v8f acc[2][2] = {};
#pragma unroll
  for (int kb = 0; kb < 16; ++kb) {
    v16bf a0 = load_afrag(ap + kb * 32);
    v16bf a1 = load_afrag(ap + (size_t)16 * EMBED + kb * 32);
    v16bf b0 = load_bfrag(bp + kb * 32);
    v16bf b1 = load_bfrag(bp + (size_t)16 * CW + kb * 32);
    acc[0][0] = wmma_acc(a0, b0, acc[0][0]);
    acc[0][1] = wmma_acc(a0, b1, acc[0][1]);
    acc[1][0] = wmma_acc(a1, b0, acc[1][0]);
    acc[1][1] = wmma_acc(a1, b1, acc[1][1]);
  }

  const float* bias = (g == 0) ? bz : (g == 1) ? br : bh;
  float* out = Ax + (size_t)g * TB * HIDDEN;
#pragma unroll
  for (int q = 0; q < 2; ++q) {
    int col = ng * 32 + q * 16 + (lane & 15);
    float bb = bias[col];
#pragma unroll
    for (int i = 0; i < 2; ++i) {
#pragma unroll
      for (int j = 0; j < 8; ++j) {
        int m = mg * 32 + i * 16 + j + ((lane >> 4) << 3);
        out[(size_t)m * HIDDEN + col] = acc[i][q][j] + bb;
      }
    }
  }
}

// ---------------- persistent GRU recurrence kernel with grid-wide barrier ----------------
#define GRID_SYNC()                                                          \
  do {                                                                       \
    __syncthreads();                                                         \
    if (threadIdx.x == 0) {                                                  \
      __threadfence();                                                       \
      unsigned g = s_gen;                                                    \
      unsigned a = atomicAdd(&bar[0], 1u);                                   \
      if (a == (unsigned)(gridDim.x - 1)) {                                  \
        atomicExch(&bar[0], 0u);                                             \
        __threadfence();                                                     \
        atomicAdd(&bar[1], 1u);                                              \
      } else {                                                               \
        while (atomicAdd(&bar[1], 0u) == g) { __builtin_amdgcn_s_sleep(2); } \
      }                                                                      \
      s_gen = g + 1u;                                                        \
      __threadfence();                                                       \
    }                                                                        \
    __syncthreads();                                                         \
  } while (0)

__global__ void k_gru(float* __restrict__ h_f32, __bf16* __restrict__ h_bf16,
                      float* __restrict__ z_f32, __bf16* __restrict__ rh_bf16,
                      __bf16* __restrict__ hs_bf16, const float* __restrict__ Ax,
                      const __bf16* __restrict__ Wg, unsigned* __restrict__ bar) {
  __shared__ unsigned s_gen;
  if (threadIdx.x == 0) s_gen = 0u;
  __syncthreads();

  int lane = threadIdx.x & 31;
  int w = blockIdx.x * 8 + (threadIdx.x >> 5);             // 0 .. 127 waves (16 blocks x 8)

  for (int t = 0; t < SEQ; ++t) {
    // ---- phase 1: P = h @ {Wz_h, Wr_h}.T ; z = sigmoid(Az+Pz), rh = sigmoid(Ar+Pr)*h
    {
      int gate = w >> 6;                                   // 0 = z, 1 = r
      int rem = w & 63;
      int mt = rem >> 5, nt = rem & 31;
      const __bf16* ap = h_bf16 + (size_t)(mt * 16 + (lane & 15)) * HIDDEN + ((lane >> 4) * 16);
      const __bf16* bp = Wg + (size_t)gate * HIDDEN * CW
                            + (size_t)(nt * 16 + (lane & 15)) * CW + EMBED + ((lane >> 4) * 16);
      v8f acc = {};
#pragma unroll
      for (int kb = 0; kb < 16; ++kb)
        acc = wmma_bf16(ap + kb * 32, bp + kb * 32, acc);

      int col = nt * 16 + (lane & 15);
      const float* Ag = Ax + ((size_t)gate * TB + (size_t)t * BATCH) * HIDDEN;
#pragma unroll
      for (int j = 0; j < 8; ++j) {
        int m = mt * 16 + j + ((lane >> 4) << 3);
        float s = sigmoidf_fast(acc[j] + Ag[(size_t)m * HIDDEN + col]);
        if (gate == 0) z_f32[m * HIDDEN + col] = s;
        else           rh_bf16[m * HIDDEN + col] = (__bf16)(s * h_f32[m * HIDDEN + col]);
      }
    }
    GRID_SYNC();
    // ---- phase 2: h_tilde = tanh(Ah + rh @ Wh_h.T); h = h + z*(h_tilde - h)
    if (w < 64) {
      int mt = w >> 5, nt = w & 31;
      const __bf16* ap = rh_bf16 + (size_t)(mt * 16 + (lane & 15)) * HIDDEN + ((lane >> 4) * 16);
      const __bf16* bp = Wg + (size_t)2 * HIDDEN * CW
                            + (size_t)(nt * 16 + (lane & 15)) * CW + EMBED + ((lane >> 4) * 16);
      v8f acc = {};
#pragma unroll
      for (int kb = 0; kb < 16; ++kb)
        acc = wmma_bf16(ap + kb * 32, bp + kb * 32, acc);

      int col = nt * 16 + (lane & 15);
      const float* Ah = Ax + ((size_t)2 * TB + (size_t)t * BATCH) * HIDDEN;
#pragma unroll
      for (int j = 0; j < 8; ++j) {
        int m = mt * 16 + j + ((lane >> 4) << 3);
        float htl = tanhf(acc[j] + Ah[(size_t)m * HIDDEN + col]);
        float hv = h_f32[m * HIDDEN + col];
        float zv = z_f32[m * HIDDEN + col];
        float hn = hv + zv * (htl - hv);
        h_f32[m * HIDDEN + col]  = hn;
        h_bf16[m * HIDDEN + col] = (__bf16)hn;
        hs_bf16[((size_t)t * BATCH + m) * HIDDEN + col] = (__bf16)hn;
      }
    }
    GRID_SYNC();
  }
}

// ---------------- output projection: logits = hs @ fc_w.T + fc_b  (M=2048,N=32000,K=512) ----
// Each wave computes a 32x64 block (2 M-tiles x 4 N-tiles) -> 8 WMMAs per 6 fragments loaded.
// Block = 128 threads (4 waves) sharing the same 32-row A panel (hits in WGP$).
// grid: x = 125 N-super-groups (4 waves x 64 cols = 256 cols each), y = 64 M-groups (32 rows).
__global__ void k_logits(const __bf16* __restrict__ hs_bf16, const __bf16* __restrict__ fcw,
                         const float* __restrict__ fc_b, float* __restrict__ out) {
  int lane = threadIdx.x & 31;
  int wv   = threadIdx.x >> 5;                             // 0..3
  int ng   = blockIdx.x * 4 + wv;                          // 0..499 (64-col groups)
  int mg   = blockIdx.y;                                   // 0..63  (32-row groups)
  int n0   = ng * 64;
  int m0   = mg * 32;

  const __bf16* ap = hs_bf16 + (size_t)(m0 + (lane & 15)) * HIDDEN + ((lane >> 4) * 16);
  const __bf16* bp = fcw     + (size_t)(n0 + (lane & 15)) * HIDDEN + ((lane >> 4) * 16);

  v8f acc[2][4] = {};
#pragma unroll
  for (int kb = 0; kb < 16; ++kb) {
    v16bf a0 = load_afrag(ap + kb * 32);
    v16bf a1 = load_afrag(ap + (size_t)16 * HIDDEN + kb * 32);
    v16bf b0 = load_bfrag(bp + kb * 32);
    v16bf b1 = load_bfrag(bp + (size_t)16 * HIDDEN + kb * 32);
    v16bf b2 = load_bfrag(bp + (size_t)32 * HIDDEN + kb * 32);
    v16bf b3 = load_bfrag(bp + (size_t)48 * HIDDEN + kb * 32);
    acc[0][0] = wmma_acc(a0, b0, acc[0][0]);
    acc[0][1] = wmma_acc(a0, b1, acc[0][1]);
    acc[0][2] = wmma_acc(a0, b2, acc[0][2]);
    acc[0][3] = wmma_acc(a0, b3, acc[0][3]);
    acc[1][0] = wmma_acc(a1, b0, acc[1][0]);
    acc[1][1] = wmma_acc(a1, b1, acc[1][1]);
    acc[1][2] = wmma_acc(a1, b2, acc[1][2]);
    acc[1][3] = wmma_acc(a1, b3, acc[1][3]);
  }

#pragma unroll
  for (int q = 0; q < 4; ++q) {
    int n = n0 + q * 16 + (lane & 15);
    float bb = fc_b[n];
#pragma unroll
    for (int i = 0; i < 2; ++i) {
#pragma unroll
      for (int j = 0; j < 8; ++j) {
        int r = m0 + i * 16 + j + ((lane >> 4) << 3);      // hs row = t*32 + b
        int tt = r >> 5, b = r & 31;
        out[((size_t)(b * SEQ + tt)) * VOCAB + n] = acc[i][q][j] + bb;  // logits[b][t][n]
      }
    }
  }
}

// ------------------------------------- launcher -------------------------------------------
extern "C" void kernel_launch(void* const* d_in, const int* in_sizes, int n_in,
                              void* d_out, int out_size, void* d_ws, size_t ws_size,
                              hipStream_t stream) {
  (void)in_sizes; (void)n_in; (void)out_size; (void)ws_size;
  const float* features  = (const float*)d_in[0];
  const int*   captions  = (const int*)  d_in[1];
  const float* embedding = (const float*)d_in[2];
  const float* Wz_w = (const float*)d_in[3];
  const float* Wz_b = (const float*)d_in[4];
  const float* Wr_w = (const float*)d_in[5];
  const float* Wr_b = (const float*)d_in[6];
  const float* Wh_w = (const float*)d_in[7];
  const float* Wh_b = (const float*)d_in[8];
  const float* fc_w = (const float*)d_in[9];
  const float* fc_b = (const float*)d_in[10];
  float* logits = (float*)d_out;

  char* ws = (char*)d_ws;
  size_t off = 0;
  auto take = [&](size_t bytes) -> char* {
    char* p = ws + off;
    off = (off + bytes + 255) & ~(size_t)255;
    return p;
  };
  unsigned* bar     = (unsigned*)take(256);
  __bf16*   x_all   = (__bf16*)  take((size_t)TB * EMBED * 2);        // 2 MB
  __bf16*   Wg      = (__bf16*)  take((size_t)3 * HIDDEN * CW * 2);   // 3 MB
  __bf16*   fcw     = (__bf16*)  take((size_t)VOCAB * HIDDEN * 2);    // 32 MB
  float*    Ax      = (float*)   take((size_t)3 * TB * HIDDEN * 4);   // 12 MB
  float*    h_f32   = (float*)   take((size_t)BATCH * HIDDEN * 4);
  __bf16*   h_bf16  = (__bf16*)  take((size_t)BATCH * HIDDEN * 2);
  float*    z_f32   = (float*)   take((size_t)BATCH * HIDDEN * 4);
  __bf16*   rh_bf16 = (__bf16*)  take((size_t)BATCH * HIDDEN * 2);
  __bf16*   hs_bf16 = (__bf16*)  take((size_t)TB * HIDDEN * 2);       // 2 MB

  // prep (also re-zeroes the grid barrier each call -> deterministic across graph replays)
  k_prep_gru<<<6144, 256, 0, stream>>>(Wz_w, Wr_w, Wh_w, Wg, bar);
  k_prep_fc <<<64000, 256, 0, stream>>>(fc_w, fcw);
  k_embed   <<<4096, 256, 0, stream>>>(captions, embedding, x_all);
  k_h0      <<<64, 256, 0, stream>>>(features, h_f32, h_bf16);

  // parallel x-part GEMM for all gates/timesteps (3072 wave jobs, 2x2 tiles each)
  k_xgemm   <<<384, 256, 0, stream>>>(x_all, Wg, Wz_b, Wr_b, Wh_b, Ax);

  // persistent recurrent kernel (16 blocks, grid-synced, 64 timesteps x 2 phases)
  k_gru     <<<16, 256, 0, stream>>>(h_f32, h_bf16, z_f32, rh_bf16, hs_bf16, Ax, Wg, bar);

  // big output GEMM: 64 M-groups x 500 N-groups, 2x4 tiles per wave
  dim3 glog(125, 64);
  k_logits  <<<glog, 128, 0, stream>>>(hs_bf16, fcw, fc_b, logits);
}
